// Block_52037823758381
// MI455X (gfx1250) — matrix operand order repaired and checked by reading, not compiled, not run on hardware
//
#include <hip/hip_runtime.h>

#define D_MODEL 2048
#define NHEAD   16
#define HS      128
#define FF_DIM  8192
#define SEQ     2048
#define BATCH   2
#define LN_EPS  1e-5f

typedef __attribute__((ext_vector_type(16))) __bf16 v16bf;
typedef __attribute__((ext_vector_type(8)))  float  v8f;

#if defined(__has_builtin)
#if __has_builtin(__builtin_amdgcn_tensor_load_to_lds) && __has_builtin(__builtin_amdgcn_s_wait_tensorcnt)
#define HAVE_TDM 1
#endif
#endif

#if defined(HAVE_TDM) && defined(__has_include)
#if __has_include(<hip/amd_detail/amd_gfx1250_TDM.h>)
#define TDM_6ARG 1   /* therock-10.0 headers -> 6-arg builtin */
#endif
#endif

typedef unsigned int v4u  __attribute__((ext_vector_type(4)));
typedef int          v8i_t __attribute__((ext_vector_type(8)));
typedef int          v4i_t __attribute__((ext_vector_type(4)));

// ---------- helpers ----------
static __device__ __forceinline__ unsigned short f2bf(float f) {
  unsigned int u = __float_as_uint(f);
  unsigned int r = (u + 0x7FFFu + ((u >> 16) & 1u)) >> 16;  // RNE
  return (unsigned short)r;
}
static __device__ __forceinline__ v8f zero8() {
  v8f z = {0.f, 0.f, 0.f, 0.f, 0.f, 0.f, 0.f, 0.f};
  return z;
}

#ifdef HAVE_TDM
static __device__ __forceinline__ unsigned lds_off(const void* p) {
  return (unsigned)(unsigned long long)(__attribute__((address_space(3))) const void*)p;
}
// 2-D bf16 tile DMA: global [tile_d1 rows x tile_d0 elems, row stride = stride0
// elements] -> LDS linear. tensor dims == tile dims (no OOB clipping needed).
static __device__ __forceinline__ void tdm_load_tile(unsigned lds_addr, const void* gptr,
                                                     unsigned tile_d0, unsigned tile_d1,
                                                     unsigned long long stride0) {
  unsigned long long ga = (unsigned long long)gptr;
  v4u g0;
  g0.x = 1u;                                                   // count=1, user D#
  g0.y = lds_addr;                                             // lds_addr [63:32]
  g0.z = (unsigned)ga;                                         // global_addr lo
  g0.w = (unsigned)((ga >> 32) & 0x01FFFFFFu) | (2u << 30);    // addr hi | type=2
  v8i_t g1;
  g1[0] = (int)(1u << 16);                                     // data_size=1 (2B)
  g1[1] = (int)((tile_d0 & 0xFFFFu) << 16);                    // tensor_dim0 lo16
  g1[2] = (int)((tile_d1 & 0xFFFFu) << 16);                    // dim0 hi=0 | tensor_dim1 lo16
  g1[3] = (int)(tile_d0 << 16);                                // dim1 hi=0 | tile_dim0
  g1[4] = (int)(tile_d1 & 0xFFFFu);                            // tile_dim1 | tile_dim2=0
  g1[5] = (int)(stride0 & 0xFFFFFFFFull);                      // dim0_stride lo32
  g1[6] = (int)((stride0 >> 32) & 0xFFFFull);                  // stride hi16 | dim1_stride=0
  g1[7] = 0;
  v4i_t z4 = {0, 0, 0, 0};
#ifdef TDM_6ARG
  v8i_t z8 = {0, 0, 0, 0, 0, 0, 0, 0};
  __builtin_amdgcn_tensor_load_to_lds(g0, g1, z4, z4, z8, 0);
#else
  __builtin_amdgcn_tensor_load_to_lds(g0, g1, z4, z4, 0);
#endif
}
static __device__ __forceinline__ void tdm_wait0() { __builtin_amdgcn_s_wait_tensorcnt(0); }
static __device__ __forceinline__ void tdm_wait2() { __builtin_amdgcn_s_wait_tensorcnt(2); }
#endif  // HAVE_TDM

// A-matrix fragment (16x32 bf16). chunk = 16 uints (32 bf16 of K).
static __device__ __forceinline__ v16bf load_fragA(const unsigned int* chunk, int hf) {
  union { v16bf v; unsigned int u[8]; } r;
#pragma unroll
  for (int i = 0; i < 4; ++i) r.u[i] = chunk[hf * 4 + i];
#pragma unroll
  for (int i = 0; i < 4; ++i) r.u[4 + i] = chunk[8 + hf * 4 + i];
  return r.v;
}
// B-matrix fragment (32x16 bf16), source staged [n][k].
static __device__ __forceinline__ v16bf load_fragB(const unsigned int* chunk, int hf) {
  union { v16bf v; unsigned int u[8]; } r;
#pragma unroll
  for (int i = 0; i < 4; ++i) r.u[i] = chunk[hf * 4 + i];
#pragma unroll
  for (int i = 0; i < 4; ++i) r.u[4 + i] = chunk[8 + hf * 4 + i];
  return r.v;
}

// ---------- weight convert + transpose: f32 [K][N] -> bf16 [N][K] ----------
__global__ __launch_bounds__(256)
void wconv_kernel(const float* __restrict__ in, unsigned short* __restrict__ out,
                  int K, int N) {
  __shared__ float tile[32][33];
  const int n0 = blockIdx.x * 32;
  const int k0 = blockIdx.y * 32;
  const int tx = threadIdx.x & 31;
  const int ty = threadIdx.x >> 5;   // 0..7
#pragma unroll
  for (int j = 0; j < 4; ++j)
    tile[ty + j * 8][tx] = in[(size_t)(k0 + ty + j * 8) * N + n0 + tx];
  __syncthreads();
#pragma unroll
  for (int j = 0; j < 4; ++j)
    out[(size_t)(n0 + ty + j * 8) * K + k0 + tx] = f2bf(tile[tx][ty + j * 8]);
}

// ---------- LayerNorm (f32 in, bf16 out): one block per row ----------
__global__ __launch_bounds__(256)
void ln_kernel(const float* __restrict__ x, const float* __restrict__ g,
               const float* __restrict__ bta, unsigned short* __restrict__ out) {
  __shared__ float red[256];
  const int row = blockIdx.x;
  const float* xr = x + (size_t)row * D_MODEL;
  float s = 0.f, ss = 0.f;
  for (int c = threadIdx.x; c < D_MODEL; c += 256) {
    float v = xr[c];
    s += v; ss += v * v;
  }
  red[threadIdx.x] = s; __syncthreads();
  for (int off = 128; off > 0; off >>= 1) {
    if (threadIdx.x < off) red[threadIdx.x] += red[threadIdx.x + off];
    __syncthreads();
  }
  const float mean = red[0] * (1.0f / D_MODEL);
  __syncthreads();
  red[threadIdx.x] = ss; __syncthreads();
  for (int off = 128; off > 0; off >>= 1) {
    if (threadIdx.x < off) red[threadIdx.x] += red[threadIdx.x + off];
    __syncthreads();
  }
  const float var = red[0] * (1.0f / D_MODEL) - mean * mean;
  const float rs = rsqrtf(var + LN_EPS);
  for (int c = threadIdx.x; c < D_MODEL; c += 256)
    out[(size_t)row * D_MODEL + c] = f2bf((xr[c] - mean) * rs * g[c] + bta[c]);
}

// ---------- BF16-WMMA GEMM, TDM double-buffered staging ----------
// out[M,N] = epi(A[M,K](bf16) @ Wt[N,K](bf16)^T + bias); flags: 1=GELU, 2=bf16 out.
__global__ __launch_bounds__(128)
void gemm_bf16_kernel(const unsigned short* __restrict__ A,
                      const unsigned short* __restrict__ Wt,
                      const float* __restrict__ bias, const float* __restrict__ res,
                      void* __restrict__ outp, int M, int K, int N, int flags) {
  __shared__ __align__(16) unsigned short As[2][64][32];
  __shared__ __align__(16) unsigned short Bs[2][64][32];
  const int tid  = threadIdx.x;
  const int wave = tid >> 5;
  const int lane = tid & 31;
  const int hf   = lane >> 4;
  const int l    = lane & 15;
  const int bm = blockIdx.y * 64;
  const int bn = blockIdx.x * 64;
  const int nsteps = K >> 5;

  v8f acc[4];
#pragma unroll
  for (int t = 0; t < 4; ++t) acc[t] = zero8();

#ifdef HAVE_TDM
  if (wave == 0) {
    tdm_load_tile(lds_off(&As[0][0][0]), A + (size_t)bm * K, 32, 64, (unsigned long long)K);
    tdm_load_tile(lds_off(&Bs[0][0][0]), Wt + (size_t)bn * K, 32, 64, (unsigned long long)K);
  }
#endif

  for (int s = 0; s < nsteps; ++s) {
    const int buf = s & 1;
#ifdef HAVE_TDM
    if (wave == 0) {
      if (s + 1 < nsteps) {
        tdm_load_tile(lds_off(&As[buf ^ 1][0][0]), A + (size_t)bm * K + (s + 1) * 32,
                      32, 64, (unsigned long long)K);
        tdm_load_tile(lds_off(&Bs[buf ^ 1][0][0]), Wt + (size_t)bn * K + (s + 1) * 32,
                      32, 64, (unsigned long long)K);
        tdm_wait2();   // the two DMAs of step s are done; step s+1 in flight
      } else {
        tdm_wait0();
      }
    }
#else
    // synchronous fallback staging (bf16 contiguous rows, uint4 chunks)
#pragma unroll
    for (int i = 0; i < 2; ++i) {
      int idx = tid + i * 128;            // 256 chunks of 16B
      int r = idx >> 2, c = idx & 3;
      *(uint4*)&As[buf][r][c * 8] =
          *(const uint4*)(A + (size_t)(bm + r) * K + s * 32 + c * 8);
      *(uint4*)&Bs[buf][r][c * 8] =
          *(const uint4*)(Wt + (size_t)(bn + r) * K + s * 32 + c * 8);
    }
#endif
    __syncthreads();

    const unsigned int* arow = (const unsigned int*)&As[buf][wave * 16 + l][0];
    v16bf afrag = load_fragA(arow, hf);
#pragma unroll
    for (int t = 0; t < 4; ++t) {
      const unsigned int* brow = (const unsigned int*)&Bs[buf][t * 16 + l][0];
      v16bf bfrag = load_fragB(brow, hf);
      acc[t] = __builtin_amdgcn_wmma_f32_16x16x32_bf16(
          false, afrag, false, bfrag, (short)0, acc[t], false, false);
    }
    __syncthreads();
  }

  const int do_gelu = flags & 1;
  const int out_bf  = flags & 2;
#pragma unroll
  for (int t = 0; t < 4; ++t) {
#pragma unroll
    for (int r = 0; r < 8; ++r) {
      int row = bm + wave * 16 + hf * 8 + r;
      int col = bn + t * 16 + l;
      float v = acc[t][r] + bias[col];
      if (do_gelu) v = 0.5f * v * (1.0f + erff(v * 0.70710678118654752f));
      if (res) v += res[(size_t)row * N + col];
      if (out_bf) ((unsigned short*)outp)[(size_t)row * N + col] = f2bf(v);
      else        ((float*)outp)[(size_t)row * N + col] = v;
    }
  }
}

// ---------- Flash attention (causal, bf16 qkv in, bf16 out) ----------
__global__ __launch_bounds__(128)
void attn_kernel(const unsigned short* __restrict__ qkv, unsigned short* __restrict__ outa) {
  __shared__ __align__(16) unsigned short Qs[64][HS];      // [q][hs]
  __shared__ __align__(16) unsigned short Ks[64][HS];      // [k][hs]
  __shared__ __align__(16) unsigned short VsT[HS][64];     // [hs][key]
  __shared__ __align__(16) unsigned short Ps[4][16][64];   // per-wave P
  const int tid  = threadIdx.x;
  const int wave = tid >> 5;
  const int lane = tid & 31;
  const int hf   = lane >> 4;
  const int l    = lane & 15;
  const int qb = blockIdx.x;
  const int h  = blockIdx.y;
  const int b  = blockIdx.z;
  const int qbase = qb * 64;
  const size_t rstride = 3 * (size_t)D_MODEL;
  const float scale = 0.088388347648318447f;   // 1/sqrt(128)

  // stage Q once
#ifdef HAVE_TDM
  if (wave == 0)
    tdm_load_tile(lds_off(&Qs[0][0]),
                  qkv + (size_t)(b * SEQ + qbase) * rstride + h * HS,
                  HS, 64, (unsigned long long)rstride);
#else
#pragma unroll
  for (int i = 0; i < 8; ++i) {
    int idx = tid + i * 128;          // 1024 chunks of 16B
    int r = idx >> 4, c = idx & 15;
    *(uint4*)&Qs[r][c * 8] =
        *(const uint4*)(qkv + (size_t)(b * SEQ + qbase + r) * rstride + h * HS + c * 8);
  }
#endif

  v8f oacc[8];
  float mrun[8], lrun[8];
#pragma unroll
  for (int t = 0; t < 8; ++t) oacc[t] = zero8();
#pragma unroll
  for (int r = 0; r < 8; ++r) { mrun[r] = -1e30f; lrun[r] = 0.f; }

  for (int jb = 0; jb <= qb; ++jb) {
    const int jbase = jb * 64;
    __syncthreads();      // all waves finished reading previous K/V
#ifdef HAVE_TDM
    if (wave == 0)
      tdm_load_tile(lds_off(&Ks[0][0]),
                    qkv + (size_t)(b * SEQ + jbase) * rstride + D_MODEL + h * HS,
                    HS, 64, (unsigned long long)rstride);
#else
#pragma unroll
    for (int i = 0; i < 8; ++i) {
      int idx = tid + i * 128;
      int r = idx >> 4, c = idx & 15;
      *(uint4*)&Ks[r][c * 8] =
          *(const uint4*)(qkv + (size_t)(b * SEQ + jbase + r) * rstride + D_MODEL + h * HS + c * 8);
    }
#endif
    // stage V transposed (manual scatter; TDM cannot transpose)
#pragma unroll
    for (int i = 0; i < 8; ++i) {
      int idx = tid + i * 128;
      int r = idx >> 4, c8 = idx & 15;
      union { uint4 q; unsigned short u[8]; } tmp;
      tmp.q = *(const uint4*)(qkv + (size_t)(b * SEQ + jbase + r) * rstride
                              + 2 * D_MODEL + h * HS + c8 * 8);
#pragma unroll
      for (int j = 0; j < 8; ++j) VsT[c8 * 8 + j][r] = tmp.u[j];
    }
#ifdef HAVE_TDM
    if (wave == 0) tdm_wait0();
#endif
    __syncthreads();

    // S = Q K^T : 4 col tiles x 4 K-chunks
    v8f sacc[4];
#pragma unroll
    for (int t = 0; t < 4; ++t) sacc[t] = zero8();
    const unsigned int* qrow = (const unsigned int*)&Qs[wave * 16 + l][0];
#pragma unroll
    for (int kc = 0; kc < 4; ++kc) {
      v16bf afrag = load_fragA(qrow + kc * 16, hf);
#pragma unroll
      for (int t = 0; t < 4; ++t) {
        const unsigned int* krow = (const unsigned int*)&Ks[t * 16 + l][0];
        v16bf bfrag = load_fragB(krow + kc * 16, hf);
        sacc[t] = __builtin_amdgcn_wmma_f32_16x16x32_bf16(
            false, afrag, false, bfrag, (short)0, sacc[t], false, false);
      }
    }

    // causal mask + online softmax (row = wave*16 + hf*8 + r)
#pragma unroll
    for (int r = 0; r < 8; ++r) {
      const int qg = qbase + wave * 16 + hf * 8 + r;
      float mx = -1e30f;
#pragma unroll
      for (int t = 0; t < 4; ++t) {
        int kg = jbase + t * 16 + l;
        float sv = sacc[t][r] * scale;
        sv = (kg <= qg) ? sv : -1e30f;
        sacc[t][r] = sv;
        mx = fmaxf(mx, sv);
      }
      for (int m = 1; m < 16; m <<= 1) mx = fmaxf(mx, __shfl_xor(mx, m, 32));
      const float mnew  = fmaxf(mrun[r], mx);
      const float alpha = __expf(mrun[r] - mnew);
      float sum = 0.f;
#pragma unroll
      for (int t = 0; t < 4; ++t) {
        float p = __expf(sacc[t][r] - mnew);
        sacc[t][r] = p;
        sum += p;
      }
      for (int m = 1; m < 16; m <<= 1) sum += __shfl_xor(sum, m, 32);
      lrun[r] = lrun[r] * alpha + sum;
      mrun[r] = mnew;
#pragma unroll
      for (int ot = 0; ot < 8; ++ot) oacc[ot][r] *= alpha;
#pragma unroll
      for (int t = 0; t < 4; ++t)
        Ps[wave][hf * 8 + r][t * 16 + l] = f2bf(sacc[t][r]);
    }

    // O += P @ V : 8 hs tiles x 2 K-chunks (keys)
    const unsigned int* prow = (const unsigned int*)&Ps[wave][l][0];
#pragma unroll
    for (int kc = 0; kc < 2; ++kc) {
      v16bf afrag = load_fragA(prow + kc * 16, hf);
#pragma unroll
      for (int ot = 0; ot < 8; ++ot) {
        const unsigned int* vrow = (const unsigned int*)&VsT[ot * 16 + l][0];
        v16bf bfrag = load_fragB(vrow + kc * 16, hf);
        oacc[ot] = __builtin_amdgcn_wmma_f32_16x16x32_bf16(
            false, afrag, false, bfrag, (short)0, oacc[ot], false, false);
      }
    }
  }

  // finalize
#pragma unroll
  for (int ot = 0; ot < 8; ++ot) {
#pragma unroll
    for (int r = 0; r < 8; ++r) {
      const int qg  = qbase + wave * 16 + hf * 8 + r;
      const int col = ot * 16 + l;
      outa[(size_t)(b * SEQ + qg) * D_MODEL + h * HS + col] = f2bf(oacc[ot][r] / lrun[r]);
    }
  }
}

// ---------- launcher ----------
extern "C" void kernel_launch(void* const* d_in, const int* in_sizes, int n_in,
                              void* d_out, int out_size, void* d_ws, size_t ws_size,
                              hipStream_t stream) {
  const float* x      = (const float*)d_in[0];
  const float* ln1_g  = (const float*)d_in[1];
  const float* ln1_b  = (const float*)d_in[2];
  const float* W_attn = (const float*)d_in[3];
  const float* b_attn = (const float*)d_in[4];
  const float* W_proj = (const float*)d_in[5];
  const float* b_proj = (const float*)d_in[6];
  const float* ln2_g  = (const float*)d_in[7];
  const float* ln2_b  = (const float*)d_in[8];
  const float* W1     = (const float*)d_in[9];
  const float* b1     = (const float*)d_in[10];
  const float* W2     = (const float*)d_in[11];
  const float* b2     = (const float*)d_in[12];
  float* out = (float*)d_out;

  char* ws = (char*)d_ws;
  typedef unsigned short u16;
  u16* Wq_t   = (u16*)(ws);                       // [6144][2048]  25165824 B
  u16* Wp_t   = (u16*)(ws + 25165824ull);         // [2048][2048]   8388608 B
  u16* W1_t   = (u16*)(ws + 33554432ull);         // [8192][2048]  33554432 B
  u16* W2_t   = (u16*)(ws + 67108864ull);         // [2048][8192]  33554432 B
  u16* h1_bf  = (u16*)(ws + 100663296ull);        // ln1 out / attn out  16777216 B
  u16* qkv_bf = (u16*)(ws + 117440512ull);        // qkv / ln2 out       50331648 B
  float* x2   = (float*)(ws + 167772160ull);      // f32 residual        33554432 B
  u16* ff1_bf = (u16*)(ws + 201326592ull);        // gelu(ffn1)          67108864 B

  const int M = BATCH * SEQ;   // 4096

  // 0) weight convert+transpose f32[K][N] -> bf16[N][K]
  wconv_kernel<<<dim3(3 * D_MODEL / 32, D_MODEL / 32), 256, 0, stream>>>(W_attn, Wq_t, D_MODEL, 3 * D_MODEL);
  wconv_kernel<<<dim3(D_MODEL / 32, D_MODEL / 32), 256, 0, stream>>>(W_proj, Wp_t, D_MODEL, D_MODEL);
  wconv_kernel<<<dim3(FF_DIM / 32, D_MODEL / 32), 256, 0, stream>>>(W1, W1_t, D_MODEL, FF_DIM);
  wconv_kernel<<<dim3(D_MODEL / 32, FF_DIM / 32), 256, 0, stream>>>(W2, W2_t, FF_DIM, D_MODEL);

  // 1) h1 = LN1(x)  (bf16)
  ln_kernel<<<M, 256, 0, stream>>>(x, ln1_g, ln1_b, h1_bf);
  // 2) qkv = h1 @ W_attn + b_attn  (bf16)
  gemm_bf16_kernel<<<dim3(3 * D_MODEL / 64, M / 64), 128, 0, stream>>>(
      h1_bf, Wq_t, b_attn, nullptr, qkv_bf, M, D_MODEL, 3 * D_MODEL, 2);
  // 3) h1 = causal attention(qkv)  (bf16)
  attn_kernel<<<dim3(SEQ / 64, NHEAD, BATCH), 128, 0, stream>>>(qkv_bf, h1_bf);
  // 4) x2 = x + h1 @ W_proj + b_proj  (f32)
  gemm_bf16_kernel<<<dim3(D_MODEL / 64, M / 64), 128, 0, stream>>>(
      h1_bf, Wp_t, b_proj, x, x2, M, D_MODEL, D_MODEL, 0);
  // 5) h2 = LN2(x2)  (bf16, reuse qkv buffer)
  ln_kernel<<<M, 256, 0, stream>>>(x2, ln2_g, ln2_b, qkv_bf);
  // 6) ff1 = gelu(h2 @ W1 + b1)  (bf16)
  gemm_bf16_kernel<<<dim3(FF_DIM / 64, M / 64), 128, 0, stream>>>(
      qkv_bf, W1_t, b1, nullptr, ff1_bf, M, D_MODEL, FF_DIM, 1 | 2);
  // 7) out = x2 + ff1 @ W2 + b2  (f32)
  gemm_bf16_kernel<<<dim3(D_MODEL / 64, M / 64), 128, 0, stream>>>(
      ff1_bf, W2_t, b2, x2, out, M, FF_DIM, D_MODEL, 0);
}
